// SelfAttention_54846732370354
// MI455X (gfx1250) — matrix operand order repaired and checked by reading, not compiled, not run on hardware
//
#include <hip/hip_runtime.h>
#include <hip/hip_bf16.h>
#include <stdint.h>

typedef __bf16 bf16_t;
typedef __bf16 v16bf __attribute__((ext_vector_type(16)));
typedef __bf16 bf16x4 __attribute__((ext_vector_type(4)));
typedef float  v8f   __attribute__((ext_vector_type(8)));

// pointer types for the async cache->LDS builtin (per hipcc diagnostic:
// param 0 is 'int __vector(4) __device__ *', i.e. int4 addrspace(1)*)
typedef int v4i_vs __attribute__((vector_size(16)));
typedef __attribute__((address_space(1))) v4i_vs* as1_v4i_p;
typedef __attribute__((address_space(3))) v4i_vs* as3_v4i_p;

#define EMB   1024
#define SEQ   2048
#define BATCH 4
#define ROWS  (BATCH * SEQ)   // 8192
#define NQKV  (3 * EMB)       // 3072

#define BM 128
#define BN 128
#define BK 32
#define NTHREADS 128          // 4 waves, 2x2, each owns a 64x64 tile

#if defined(__has_builtin)
#if __has_builtin(__builtin_amdgcn_global_load_async_to_lds_b128)
#define HAVE_ASYNC_LDS 1
#endif
#if __has_builtin(__builtin_amdgcn_s_wait_asynccnt)
#define HAVE_WAIT_ASYNC 1
#endif
#endif
#ifndef HAVE_ASYNC_LDS
#define HAVE_ASYNC_LDS 0
#endif
#ifndef HAVE_WAIT_ASYNC
#define HAVE_WAIT_ASYNC 0
#endif

__device__ __forceinline__ bf16_t f2bf(float f) { return (bf16_t)f; }

__device__ __forceinline__ void wait_async0() {
#if HAVE_ASYNC_LDS
#if HAVE_WAIT_ASYNC
  __builtin_amdgcn_s_wait_asynccnt(0);
#else
  asm volatile("s_wait_asynccnt 0x0" ::: "memory");
#endif
#endif
}

// one 16-byte cache->LDS async copy (no VGPR data, tracked by ASYNCcnt)
__device__ __forceinline__ void cp16_g2l(const bf16_t* g, bf16_t* l) {
#if HAVE_ASYNC_LDS
  __builtin_amdgcn_global_load_async_to_lds_b128((as1_v4i_p)g, (as3_v4i_p)l, 0, 0);
#else
  *reinterpret_cast<uint4*>(l) = *reinterpret_cast<const uint4*>(g);
#endif
}

// ---------------------------------------------------------------- fp32 -> bf16
__global__ void k_cvt_bf16(const float* __restrict__ src,
                           bf16_t* __restrict__ dst, int n4) {
  int i = blockIdx.x * blockDim.x + threadIdx.x;
  if (i >= n4) return;
  float4 v = reinterpret_cast<const float4*>(src)[i];
  bf16x4 o;
  o.x = f2bf(v.x); o.y = f2bf(v.y); o.z = f2bf(v.z); o.w = f2bf(v.w);
  reinterpret_cast<bf16x4*>(dst)[i] = o;
}

// ------------------------------------------------- w [K][N] -> wT [N][K] bf16
__global__ void k_transpose_bf16(const float* __restrict__ src,
                                 bf16_t* __restrict__ dst, int K, int N) {
  __shared__ float tile[32][33];
  int n0 = blockIdx.x * 32, k0 = blockIdx.y * 32;
  int tx = threadIdx.x, ty = threadIdx.y;
  for (int i = ty; i < 32; i += 8)
    tile[i][tx] = src[(size_t)(k0 + i) * N + n0 + tx];
  __syncthreads();
  for (int i = ty; i < 32; i += 8)
    dst[(size_t)(n0 + i) * K + k0 + tx] = f2bf(tile[tx][i]);
}

// --------------------------------------------------------------- GEMM core
// C[M,N] = A[M,K] * Bt[N,K]^T ; MODE 0: +bias -> qk/vT ; 1: sigmoid -> P ; 2: f32 out
template <int MODE>
__global__ void __launch_bounds__(NTHREADS, 1)
k_gemm(const bf16_t* __restrict__ A, int lda, size_t sA,
       const bf16_t* __restrict__ Bt, int ldb, size_t sB,
       int K,
       const float* __restrict__ bias,
       bf16_t* __restrict__ obf, size_t sObf,
       bf16_t* __restrict__ vT,
       float* __restrict__ of32, size_t sOf) {
  __shared__ __align__(16) bf16_t As[2][BM][BK];
  __shared__ __align__(16) bf16_t Bs[2][BN][BK];

  const int z = blockIdx.z;
  A  += (size_t)z * sA;
  Bt += (size_t)z * sB;

  const int m0 = blockIdx.y * BM;
  const int n0 = blockIdx.x * BN;
  const int tid  = threadIdx.x;
  const int lane = tid & 31;
  const int r    = lane & 15;
  const int half = lane >> 4;
  const int wid  = tid >> 5;
  const int wM   = wid & 1;   // 2 waves along M (64 rows each)
  const int wN   = wid >> 1;  // 2 waves along N (64 cols each)

  const v8f vzero = {0.f, 0.f, 0.f, 0.f, 0.f, 0.f, 0.f, 0.f};
  v8f acc[4][4];
#pragma unroll
  for (int i = 0; i < 4; ++i)
#pragma unroll
    for (int j = 0; j < 4; ++j) acc[i][j] = vzero;

  const int KT = K / BK;

  // stage K-tile kt into LDS buffer nb: 512 16B chunks each for A and B,
  // 4 + 4 chunks per thread, all via async cache->LDS copies.
#define STAGE_TILE(kt_, nb_)                                                   \
  {                                                                            \
    const int k0_ = (kt_) * BK;                                                \
    _Pragma("unroll") for (int it = 0; it < 4; ++it) {                         \
      int c = tid + it * NTHREADS;                                             \
      int rr = c >> 2, cc = (c & 3) * 8;                                       \
      cp16_g2l(A + (size_t)(m0 + rr) * lda + k0_ + cc, &As[nb_][rr][cc]);      \
      cp16_g2l(Bt + (size_t)(n0 + rr) * ldb + k0_ + cc, &Bs[nb_][rr][cc]);     \
    }                                                                          \
  }

  STAGE_TILE(0, 0);
  wait_async0();
  __syncthreads();

  for (int kt = 0; kt < KT; ++kt) {
    const int buf = kt & 1;
    const bool nxt = (kt + 1) < KT;
    if (nxt) STAGE_TILE(kt + 1, buf ^ 1);   // overlaps with WMMA below

    // A fragments: row r, K chunks [8h,+8) and [16+8h,+8)
    v16bf af[4];
#pragma unroll
    for (int im = 0; im < 4; ++im) {
      const bf16_t* p = &As[buf][wM * 64 + im * 16 + r][0];
      union { uint4 u[2]; v16bf v; } x;
      x.u[0] = *reinterpret_cast<const uint4*>(p + 8 * half);
      x.u[1] = *reinterpret_cast<const uint4*>(p + 16 + 8 * half);
      af[im] = x.v;
    }
#pragma unroll
    for (int jn = 0; jn < 4; ++jn) {
      // B fragment: row n, K chunks [8h,+8) and [16+8h,+8)
      const bf16_t* p = &Bs[buf][wN * 64 + jn * 16 + r][0];
      union { uint4 u[2]; v16bf v; } x;
      x.u[0] = *reinterpret_cast<const uint4*>(p + 8 * half);
      x.u[1] = *reinterpret_cast<const uint4*>(p + 16 + 8 * half);
      v16bf bfrag = x.v;
#pragma unroll
      for (int im = 0; im < 4; ++im) {
        acc[im][jn] = __builtin_amdgcn_wmma_f32_16x16x32_bf16(
            false, af[im], false, bfrag, (short)0, acc[im][jn], false, false);
      }
    }

    if (nxt) wait_async0();   // next tile fully in LDS before barrier
    __syncthreads();
  }
#undef STAGE_TILE

  // epilogue: C layout row = i + 8*half, col = lane&15
#pragma unroll
  for (int jn = 0; jn < 4; ++jn) {
    const int col = n0 + wN * 64 + jn * 16 + r;
    float bv = 0.f;
    if (MODE == 0) bv = bias[col];
#pragma unroll
    for (int im = 0; im < 4; ++im) {
      const int rowb = m0 + wM * 64 + im * 16 + half * 8;
      v8f a = acc[im][jn];
#pragma unroll
      for (int i = 0; i < 8; ++i) {
        const int row = rowb + i;
        float val = a[i];
        if (MODE == 0) {
          val += bv;
          if (col < 2 * EMB) {                       // q,k interleaved rows
            obf[(size_t)row * (2 * EMB) + col] = f2bf(val);
          } else {                                   // v stored transposed
            const int b = row >> 11, s = row & (SEQ - 1), d = col - 2 * EMB;
            vT[((size_t)b * EMB + d) * SEQ + s] = f2bf(val);
          }
        } else if (MODE == 1) {
          float sc = val * 32.0f;                    // / D^-0.5 == * sqrt(1024)
          float pr = 1.0f / (1.0f + __expf(-sc));
          obf[(size_t)z * sObf + (size_t)row * SEQ + col] = f2bf(pr);
        } else {
          of32[(size_t)z * sOf + (size_t)row * EMB + col] = val;
        }
      }
    }
  }
}

// ------------------------------------------------------------------ launcher
extern "C" void kernel_launch(void* const* d_in, const int* in_sizes, int n_in,
                              void* d_out, int out_size, void* d_ws, size_t ws_size,
                              hipStream_t stream) {
  const float* x = (const float*)d_in[0];
  const float* w = (const float*)d_in[1];
  const float* b = (const float*)d_in[2];
  float* out = (float*)d_out;

  char* ws = (char*)d_ws;
  size_t off = 0;
  bf16_t* x_bf = (bf16_t*)(ws + off); off += (size_t)ROWS * EMB * 2;        // 16 MB
  bf16_t* wT   = (bf16_t*)(ws + off); off += (size_t)NQKV * EMB * 2;        //  6 MB
  bf16_t* qk   = (bf16_t*)(ws + off); off += (size_t)ROWS * (2 * EMB) * 2;  // 32 MB
  bf16_t* vT   = (bf16_t*)(ws + off); off += (size_t)BATCH * EMB * SEQ * 2; // 16 MB
  bf16_t* P    = (bf16_t*)(ws + off);                                       // 32 MB

  {  // x -> bf16
    int n4 = ROWS * EMB / 4;
    k_cvt_bf16<<<(n4 + 255) / 256, 256, 0, stream>>>(x, x_bf, n4);
  }
  {  // w [1024][3072] -> wT [3072][1024] bf16
    dim3 g(NQKV / 32, EMB / 32), blk(32, 8);
    k_transpose_bf16<<<g, blk, 0, stream>>>(w, wT, EMB, NQKV);
  }
  {  // qkv = x @ w + b   (M=8192, N=3072, K=1024)
    dim3 g(NQKV / BN, ROWS / BM, 1);
    k_gemm<0><<<g, NTHREADS, 0, stream>>>(x_bf, EMB, 0, wT, EMB, 0, EMB,
                                          b, qk, 0, vT, nullptr, 0);
  }
  {  // P = sigmoid(32 * q @ k^T)   per batch (M=N=2048, K=1024)
    dim3 g(SEQ / BN, SEQ / BM, BATCH);
    size_t sAB = (size_t)SEQ * (2 * EMB);
    k_gemm<1><<<g, NTHREADS, 0, stream>>>(qk, 2 * EMB, sAB,
                                          qk + EMB, 2 * EMB, sAB, EMB,
                                          nullptr, P, (size_t)SEQ * SEQ,
                                          nullptr, nullptr, 0);
  }
  {  // out = P @ v   per batch (M=2048, N=1024, K=2048)
    dim3 g(EMB / BN, SEQ / BM, BATCH);
    k_gemm<2><<<g, NTHREADS, 0, stream>>>(P, SEQ, (size_t)SEQ * SEQ,
                                          vT, SEQ, (size_t)EMB * SEQ, SEQ,
                                          nullptr, nullptr, 0, nullptr,
                                          out, (size_t)SEQ * EMB);
  }
}
